// Attention_21801253995050
// MI455X (gfx1250) — matrix operand order repaired and checked by reading, not compiled, not run on hardware
//
#include <hip/hip_runtime.h>
#include <hip/hip_bf16.h>

// ---------------------------------------------------------------------------
// GPT self-attention block for MI455X (gfx1250), bf16 WMMA pipeline.
// B=4, S=2048, D=1024, H=16, hd=64.
// ---------------------------------------------------------------------------

typedef __attribute__((ext_vector_type(16))) __bf16 v16bf;
typedef __attribute__((ext_vector_type(8)))  float  v8f;

#define WMMA_BF16(a, b, c) \
  __builtin_amdgcn_wmma_f32_16x16x32_bf16(false, (a), false, (b), (short)0, (c), false, false)

__device__ inline unsigned short f2bf(float f) {
  union { float f; unsigned u; } a; a.f = f;
  unsigned u = a.u;
  u += 0x7fffu + ((u >> 16) & 1u);   // round-to-nearest-even
  return (unsigned short)(u >> 16);
}

__device__ inline v8f zero8() {
  v8f z = {0.f, 0.f, 0.f, 0.f, 0.f, 0.f, 0.f, 0.f};
  return z;
}

union FragU { v16bf v; unsigned u[8]; };

// A-fragment (16xK slice, K window of 32 bf16) from LDS tile stored row-major
// [m][k], k contiguous. strideU = row stride in uints (bf16 pairs).
// Contiguous k order (k = 16h + s), same as B: lane m/m+16 holds row m; uint j -> Kpair j+8h.
__device__ inline v16bf load_frag_a(const unsigned* lds, int mBase, int kOffU, int strideU) {
  const int lane = threadIdx.x & 31;
  const int m = lane & 15, h = lane >> 4;
  const unsigned* row = lds + (mBase + m) * strideU + kOffU;
  FragU f;
#pragma unroll
  for (int j = 0; j < 4; ++j) {
    f.u[j]     = row[j + 8 * h];
    f.u[j + 4] = row[4 + j + 8 * h];
  }
  return f.v;
}

// B-fragment (Kx16 slice, K window of 32 bf16) from LDS tile stored [n][k],
// k contiguous. Contiguous k order (k = 16h + s), same as A: lane n/n+16 holds col n; uint v -> Kpair v+8h.
__device__ inline v16bf load_frag_b(const unsigned* lds, int nBase, int kOffU, int strideU) {
  const int lane = threadIdx.x & 31;
  const int n = lane & 15, h = lane >> 4;
  const unsigned* row = lds + (nBase + n) * strideU + kOffU;
  FragU f;
#pragma unroll
  for (int j = 0; j < 8; ++j) f.u[j] = row[j + 8 * h];
  return f.v;
}

// ---------------------------------------------------------------------------
// Stage 0: fp32 -> bf16 conversion / weight transposition
// ---------------------------------------------------------------------------
__global__ void convert_bf(const float* __restrict__ in, unsigned short* __restrict__ out, int n) {
  int i = (blockIdx.x * blockDim.x + threadIdx.x) * 4;
#pragma unroll
  for (int j = 0; j < 4; ++j)
    if (i + j < n) out[i + j] = f2bf(in[i + j]);
}

// in: [K][N] fp32, out: [N][K] bf16 (k contiguous -> matches B-fragment wants)
__global__ void transpose_bf(const float* __restrict__ in, unsigned short* __restrict__ out,
                             int K, int N) {
  int idx = blockIdx.x * blockDim.x + threadIdx.x;
  if (idx >= K * N) return;
  int n = idx / K, k = idx - n * K;
  out[idx] = f2bf(in[k * N + n]);
}

// ---------------------------------------------------------------------------
// Stage 1: QKV GEMM  [8192 x 3072] = Xbf[8192 x 1024] * W[1024 x 3072] + b
// 128x128 block tile, 8 waves (4M x 2N), each wave 32x64 via 2x4 WMMA accs.
// Epilogue scatters into head-major Q [B,H,S,hd] (pre-scaled by 1/8),
// K [B,H,S,hd], and transposed V [B,H,hd,S].
// ---------------------------------------------------------------------------
__global__ __launch_bounds__(256) void qkv_gemm(
    const unsigned* __restrict__ Xu,    // [8192][512] uints (bf16 pairs)
    const unsigned* __restrict__ Wtu,   // [3072][512] uints (W^T bf16)
    const float* __restrict__ bias,     // [3072]
    unsigned short* __restrict__ Qb,    // [64][2048][64]
    unsigned short* __restrict__ Kb,    // [64][2048][64]
    unsigned short* __restrict__ Vt)    // [64][64][2048]
{
  __shared__ unsigned sA[128 * 16];
  __shared__ unsigned sB[128 * 16];

  const int t = threadIdx.x;
  const int wave = t >> 5, lane = t & 31;
  const int waveM = wave >> 1, waveN = wave & 1;
  const int m0 = blockIdx.y * 128, n0 = blockIdx.x * 128;
  const int ar = t >> 1, ac = (t & 1) * 8;

  v8f acc[2][4];
#pragma unroll
  for (int i = 0; i < 2; ++i)
#pragma unroll
    for (int j = 0; j < 4; ++j) acc[i][j] = zero8();

  for (int kk = 0; kk < 512; kk += 16) {   // K loop, 32 bf16 (=16 uints) per step
    __syncthreads();
    const unsigned* gA = Xu  + (size_t)(m0 + ar) * 512 + kk + ac;
    const unsigned* gB = Wtu + (size_t)(n0 + ar) * 512 + kk + ac;
    __builtin_prefetch(gA + 16, 0, 0);
    __builtin_prefetch(gB + 16, 0, 0);
#pragma unroll
    for (int i = 0; i < 8; ++i) {
      sA[ar * 16 + ac + i] = gA[i];
      sB[ar * 16 + ac + i] = gB[i];
    }
    __syncthreads();

    v16bf a0 = load_frag_a(sA, waveM * 32,      0, 16);
    v16bf a1 = load_frag_a(sA, waveM * 32 + 16, 0, 16);
#pragma unroll
    for (int n = 0; n < 4; ++n) {
      v16bf bb = load_frag_b(sB, waveN * 64 + n * 16, 0, 16);
      acc[0][n] = WMMA_BF16(a0, bb, acc[0][n]);
      acc[1][n] = WMMA_BF16(a1, bb, acc[1][n]);
    }
  }

  const int rowHi = lane >> 4, nLane = lane & 15;
#pragma unroll
  for (int mi = 0; mi < 2; ++mi)
#pragma unroll
    for (int n = 0; n < 4; ++n)
#pragma unroll
      for (int r = 0; r < 8; ++r) {
        int mg = m0 + waveM * 32 + mi * 16 + r + 8 * rowHi;
        int ng = n0 + waveN * 64 + n * 16 + nLane;
        float v = acc[mi][n][r] + bias[ng];
        int b = mg >> 11, s = mg & 2047;
        int which = ng >> 10, c = ng & 1023;
        int hh = c >> 6, d = c & 63;
        int bh = b * 16 + hh;
        if (which == 0)      Qb[((size_t)bh * 2048 + s) * 64 + d] = f2bf(v * 0.125f);
        else if (which == 1) Kb[((size_t)bh * 2048 + s) * 64 + d] = f2bf(v);
        else                 Vt[((size_t)bh * 64 + d) * 2048 + s] = f2bf(v);
      }
}

// ---------------------------------------------------------------------------
// Stage 2: causal flash attention. One block per (b, h, 64-query tile);
// 4 waves, each owns 16 query rows. Online softmax; WMMA for QK^T and PV.
// ---------------------------------------------------------------------------
__global__ __launch_bounds__(128) void flash_attn(
    const unsigned short* __restrict__ Qb,   // [64][2048][64] (pre-scaled)
    const unsigned short* __restrict__ Kb,   // [64][2048][64]
    const unsigned short* __restrict__ Vt,   // [64][64][2048]
    unsigned short* __restrict__ Ob)         // [4][2048][1024] merged heads
{
  __shared__ unsigned sQ[64 * 32];        // [q][d]  (A tiles)
  __shared__ unsigned sK[64 * 32];        // [key][d] (B tiles for QK^T)
  __shared__ unsigned sV[64 * 32];        // [d][key] (B tiles for PV)
  __shared__ unsigned short sP[64 * 64];  // [q][key] (A tiles for PV)

  const int t = threadIdx.x;
  const int wave = t >> 5, lane = t & 31;
  const int rowHi = lane >> 4, nLane = lane & 15;
  const int q0 = blockIdx.x * 64;
  const int h = blockIdx.y, b = blockIdx.z;
  const int bh = b * 16 + h;

  const unsigned* Qu = (const unsigned*)Qb + (size_t)bh * 2048 * 32;
  const unsigned* Ku = (const unsigned*)Kb + (size_t)bh * 2048 * 32;
  const unsigned* Vu = (const unsigned*)Vt + (size_t)bh * 64 * 1024;

  {
    const int r = t >> 1, cb = (t & 1) * 16;
#pragma unroll
    for (int i = 0; i < 16; ++i)
      sQ[r * 32 + cb + i] = Qu[(size_t)(q0 + r) * 32 + cb + i];
  }

  v8f o[4];
#pragma unroll
  for (int n = 0; n < 4; ++n) o[n] = zero8();
  float m_i[8], l_i[8];
#pragma unroll
  for (int r = 0; r < 8; ++r) { m_i[r] = -1e30f; l_i[r] = 0.f; }

  for (int kb = 0; kb <= q0; kb += 64) {
    __syncthreads();   // previous iteration done reading sK/sV/sP
    {
      const int r = t >> 1, cb = (t & 1) * 16;
#pragma unroll
      for (int i = 0; i < 16; ++i) {
        sK[r * 32 + cb + i] = Ku[(size_t)(kb + r) * 32 + cb + i];
        sV[r * 32 + cb + i] = Vu[(size_t)r * 1024 + (kb >> 1) + cb + i];
      }
    }
    __syncthreads();

    // S = Q * K^T  (hd=64 -> 2 K-steps of 32)
    v8f sc[4];
#pragma unroll
    for (int n = 0; n < 4; ++n) sc[n] = zero8();
#pragma unroll
    for (int ks = 0; ks < 2; ++ks) {
      v16bf aq = load_frag_a(sQ, wave * 16, ks * 16, 32);
#pragma unroll
      for (int n = 0; n < 4; ++n) {
        v16bf bk = load_frag_b(sK, n * 16, ks * 16, 32);
        sc[n] = WMMA_BF16(aq, bk, sc[n]);
      }
    }

    // causal mask (analytic; ignores the dense bool mask input)
#pragma unroll
    for (int n = 0; n < 4; ++n) {
      int key = kb + n * 16 + nLane;
#pragma unroll
      for (int r = 0; r < 8; ++r) {
        int q = q0 + wave * 16 + r + 8 * rowHi;
        if (key > q) sc[n][r] = -1e30f;
      }
    }

    // online softmax: row stats live per C-layout row r (+8*rowHi half)
    float corr[8];
#pragma unroll
    for (int r = 0; r < 8; ++r) {
      float tmax = fmaxf(fmaxf(sc[0][r], sc[1][r]), fmaxf(sc[2][r], sc[3][r]));
#pragma unroll
      for (int xm = 1; xm < 16; xm <<= 1) tmax = fmaxf(tmax, __shfl_xor(tmax, xm, 32));
      float nm = fmaxf(m_i[r], tmax);
      corr[r] = __expf(m_i[r] - nm);
      m_i[r] = nm;
    }
#pragma unroll
    for (int n = 0; n < 4; ++n)
#pragma unroll
      for (int r = 0; r < 8; ++r)
        sc[n][r] = __expf(sc[n][r] - m_i[r]);
#pragma unroll
    for (int r = 0; r < 8; ++r) {
      float s = sc[0][r] + sc[1][r] + sc[2][r] + sc[3][r];
#pragma unroll
      for (int xm = 1; xm < 16; xm <<= 1) s += __shfl_xor(s, xm, 32);
      l_i[r] = l_i[r] * corr[r] + s;
    }
#pragma unroll
    for (int n = 0; n < 4; ++n)
#pragma unroll
      for (int r = 0; r < 8; ++r)
        o[n][r] *= corr[r];

    // P: C-layout regs -> LDS row-major so it can re-enter as an A fragment
#pragma unroll
    for (int n = 0; n < 4; ++n)
#pragma unroll
      for (int r = 0; r < 8; ++r) {
        int m = wave * 16 + r + 8 * rowHi;
        sP[m * 64 + n * 16 + nLane] = f2bf(sc[n][r]);
      }
    __syncthreads();

    // O += P * V  (keys=64 -> 2 K-steps of 32)
#pragma unroll
    for (int ks = 0; ks < 2; ++ks) {
      v16bf ap = load_frag_a((const unsigned*)sP, wave * 16, ks * 16, 32);
#pragma unroll
      for (int n = 0; n < 4; ++n) {
        v16bf bv = load_frag_b(sV, n * 16, ks * 16, 32);
        o[n] = WMMA_BF16(ap, bv, o[n]);
      }
    }
  }

  // normalize and write merged-head output (bf16)
#pragma unroll
  for (int n = 0; n < 4; ++n)
#pragma unroll
    for (int r = 0; r < 8; ++r) {
      int s = q0 + wave * 16 + r + 8 * rowHi;
      int d = n * 16 + nLane;
      float val = o[n][r] / l_i[r];
      Ob[((size_t)b * 2048 + s) * 1024 + h * 64 + d] = f2bf(val);
    }
}

// ---------------------------------------------------------------------------
// Stage 3: output projection  out[8192 x 1024] = O * c_proj_w + b  (fp32 out)
// ---------------------------------------------------------------------------
__global__ __launch_bounds__(256) void proj_gemm(
    const unsigned* __restrict__ Ou,    // [8192][512]
    const unsigned* __restrict__ Ptu,   // [1024][512]  (c_proj_w^T bf16)
    const float* __restrict__ bias,     // [1024]
    float* __restrict__ out)            // [8192][1024] fp32
{
  __shared__ unsigned sA[128 * 16];
  __shared__ unsigned sB[128 * 16];

  const int t = threadIdx.x;
  const int wave = t >> 5, lane = t & 31;
  const int waveM = wave >> 1, waveN = wave & 1;
  const int m0 = blockIdx.y * 128, n0 = blockIdx.x * 128;
  const int ar = t >> 1, ac = (t & 1) * 8;

  v8f acc[2][4];
#pragma unroll
  for (int i = 0; i < 2; ++i)
#pragma unroll
    for (int j = 0; j < 4; ++j) acc[i][j] = zero8();

  for (int kk = 0; kk < 512; kk += 16) {
    __syncthreads();
    const unsigned* gA = Ou  + (size_t)(m0 + ar) * 512 + kk + ac;
    const unsigned* gB = Ptu + (size_t)(n0 + ar) * 512 + kk + ac;
    __builtin_prefetch(gA + 16, 0, 0);
    __builtin_prefetch(gB + 16, 0, 0);
#pragma unroll
    for (int i = 0; i < 8; ++i) {
      sA[ar * 16 + ac + i] = gA[i];
      sB[ar * 16 + ac + i] = gB[i];
    }
    __syncthreads();

    v16bf a0 = load_frag_a(sA, waveM * 32,      0, 16);
    v16bf a1 = load_frag_a(sA, waveM * 32 + 16, 0, 16);
#pragma unroll
    for (int n = 0; n < 4; ++n) {
      v16bf bb = load_frag_b(sB, waveN * 64 + n * 16, 0, 16);
      acc[0][n] = WMMA_BF16(a0, bb, acc[0][n]);
      acc[1][n] = WMMA_BF16(a1, bb, acc[1][n]);
    }
  }

  const int rowHi = lane >> 4, nLane = lane & 15;
#pragma unroll
  for (int mi = 0; mi < 2; ++mi)
#pragma unroll
    for (int n = 0; n < 4; ++n)
#pragma unroll
      for (int r = 0; r < 8; ++r) {
        int mg = m0 + waveM * 32 + mi * 16 + r + 8 * rowHi;
        int ng = n0 + waveN * 64 + n * 16 + nLane;
        out[(size_t)mg * 1024 + ng] = acc[mi][n][r] + bias[ng];
      }
}

// ---------------------------------------------------------------------------
// Host launcher
// ---------------------------------------------------------------------------
extern "C" void kernel_launch(void* const* d_in, const int* in_sizes, int n_in,
                              void* d_out, int out_size, void* d_ws, size_t ws_size,
                              hipStream_t stream) {
  (void)in_sizes; (void)n_in; (void)out_size; (void)ws_size;

  const float* x        = (const float*)d_in[0];  // [4,2048,1024]
  const float* c_attn_w = (const float*)d_in[1];  // [1024,3072]
  const float* c_attn_b = (const float*)d_in[2];  // [3072]
  const float* c_proj_w = (const float*)d_in[3];  // [1024,1024]
  const float* c_proj_b = (const float*)d_in[4];  // [1024]
  // d_in[5] = dense causal mask: unused (mask computed analytically)

  char* ws = (char*)d_ws;
  size_t off = 0;
  auto alloc = [&](size_t bytes) -> void* {
    void* p = ws + off;
    off += (bytes + 255) & ~(size_t)255;
    return p;
  };

  unsigned short* Xbf = (unsigned short*)alloc((size_t)8192 * 1024 * 2);  // x bf16
  unsigned short* Wt  = (unsigned short*)alloc((size_t)3072 * 1024 * 2);  // c_attn_w^T bf16
  unsigned short* Pt  = (unsigned short*)alloc((size_t)1024 * 1024 * 2);  // c_proj_w^T bf16
  unsigned short* Qb  = (unsigned short*)alloc((size_t)64 * 2048 * 64 * 2);
  unsigned short* Kb  = (unsigned short*)alloc((size_t)64 * 2048 * 64 * 2);
  unsigned short* Vt  = (unsigned short*)alloc((size_t)64 * 64 * 2048 * 2);
  unsigned short* Ob  = (unsigned short*)alloc((size_t)8192 * 1024 * 2);

  // Stage 0: precision conversion (bandwidth-bound, trivially parallel)
  convert_bf<<<8192, 256, 0, stream>>>(x, Xbf, 8192 * 1024);
  transpose_bf<<<(3072 * 1024 + 255) / 256, 256, 0, stream>>>(c_attn_w, Wt, 1024, 3072);
  transpose_bf<<<(1024 * 1024 + 255) / 256, 256, 0, stream>>>(c_proj_w, Pt, 1024, 1024);

  // Stage 1: QKV GEMM (WMMA)
  qkv_gemm<<<dim3(24, 64), 256, 0, stream>>>(
      (const unsigned*)Xbf, (const unsigned*)Wt, c_attn_b, Qb, Kb, Vt);

  // Stage 2: causal flash attention (WMMA)
  flash_attn<<<dim3(32, 16, 4), 128, 0, stream>>>(Qb, Kb, Vt, Ob);

  // Stage 3: output projection (WMMA, fp32 out)
  proj_gemm<<<dim3(8, 64), 256, 0, stream>>>(
      (const unsigned*)Ob, (const unsigned*)Pt, c_proj_b, (float*)d_out);
}
